// CrossAttention_87101936763227
// MI455X (gfx1250) — matrix operand order repaired and checked by reading, not compiled, not run on hardware
//
#include <hip/hip_runtime.h>

// ---------------------------------------------------------------------------
// CrossAttention (B=2, S=4096, D=512, H=8, dh=64) for gfx1250 (MI455X).
// bf16 WMMA everywhere: v_wmma_f32_16x16x32_bf16, f32 accumulation,
// flash-attention online softmax so the 4096x4096 score matrices never
// touch memory. Projection GEMMs use 32x64 tiles per wave for 1.5 b128 loads
// per WMMA (vs 2.5 with 16x64 tiles).
// ---------------------------------------------------------------------------

#define S_LEN   4096
#define NROW    8192          // B*S
#define DMODEL  512
#define DH      64
#define NBH     16            // B*H

#define DEVINL static __device__ __forceinline__

typedef __attribute__((ext_vector_type(16))) __bf16 v16bf;
typedef __attribute__((ext_vector_type(8)))  float  v8f;

union FragBF { uint4 q[2]; v16bf v; };

DEVINL unsigned short f2bf(float f) {
  unsigned int u = __float_as_uint(f);
  u += 0x7FFFu + ((u >> 16) & 1u);          // round-to-nearest-even
  return (unsigned short)(u >> 16);
}

// A-matrix fragment (16xK slab, 16-bit): lane L holds row M=L%16, group g=L/16.
// Halves: K = k0 + g*8 + [0..7]  and  K = k0 + 16 + g*8 + [0..7]  (ISA 7.12.2).
DEVINL v16bf load_a_frag(const unsigned short* base, int ld, int m0, int k0, int lane) {
  const int M = lane & 15, g = (lane >> 4) & 1;
  const unsigned short* p = base + (size_t)(m0 + M) * (size_t)ld + k0 + g * 8;
  FragBF f;
  f.q[0] = *(const uint4*)(p);
  f.q[1] = *(const uint4*)(p + 16);
  return f.v;
}

// B-matrix fragment (32x16, 16-bit): lane L holds column N=L%16, group g=L/16.
// Halves: K = k0 + g*8 + [0..7]  and  K = k0 + 16 + g*8 + [0..7]. Source stored [N, K] row-major.
DEVINL v16bf load_b_frag(const unsigned short* base, int ld, int n0, int k0, int lane) {
  const int N = lane & 15, g = (lane >> 4) & 1;
  const unsigned short* p = base + (size_t)(n0 + N) * (size_t)ld + k0 + g * 8;
  FragBF f;
  f.q[0] = *(const uint4*)(p);
  f.q[1] = *(const uint4*)(p + 16);
  return f.v;
}

DEVINL v8f wmma_bf16(v16bf a, v16bf b, v8f c) {
  return __builtin_amdgcn_wmma_f32_16x16x32_bf16(false, a, false, b,
                                                 (short)0, c, false, false);
}

// ---------------------------------------------------------------------------
// Kernel 0: f32 -> bf16 downconvert (vectorized x4)
// ---------------------------------------------------------------------------
__global__ __launch_bounds__(256) void cvt_bf16_kernel(
    const float* __restrict__ src, unsigned short* __restrict__ dst, int n4) {
  int i = blockIdx.x * blockDim.x + threadIdx.x;
  if (i < n4) {
    float4 f = ((const float4*)src)[i];
    ushort4 o;
    o.x = f2bf(f.x); o.y = f2bf(f.y); o.z = f2bf(f.z); o.w = f2bf(f.w);
    ((ushort4*)dst)[i] = o;
  }
}

// ---------------------------------------------------------------------------
// Kernel 1: QKV projection. grid = (DMODEL/64, NROW/128, 3), block = 128.
// Each wave: 32 (M) x 64 (N) tile -> 8 WMMA per 32-wide k-step.
// Y[n,e] = sum_d X[n,d] * W[e,d].  Writes Q,K as [BH,S,dh] and V as [BH,dh,S].
// ---------------------------------------------------------------------------
__global__ __launch_bounds__(128) void qkv_gemm_kernel(
    const unsigned short* __restrict__ Xbf,
    const unsigned short* __restrict__ Wq,
    const unsigned short* __restrict__ Wk,
    const unsigned short* __restrict__ Wv,
    unsigned short* __restrict__ Qws,
    unsigned short* __restrict__ Kws,
    unsigned short* __restrict__ Vtws) {
  const int lane = threadIdx.x & 31;
  const int wave = threadIdx.x >> 5;
  const int n0   = blockIdx.x * 64;                // output column tile (e)
  const int m0   = blockIdx.y * 128 + wave * 32;   // output row tile (n)
  const int z    = blockIdx.z;
  const unsigned short* W = (z == 0) ? Wq : (z == 1) ? Wk : Wv;

  v8f acc[2][4];
#pragma unroll
  for (int u = 0; u < 2; ++u)
#pragma unroll
    for (int t = 0; t < 4; ++t)
#pragma unroll
      for (int r = 0; r < 8; ++r) acc[u][t][r] = 0.0f;

  for (int k0 = 0; k0 < DMODEL; k0 += 32) {
    v16bf a0 = load_a_frag(Xbf, DMODEL, m0, k0, lane);
    v16bf a1 = load_a_frag(Xbf, DMODEL, m0 + 16, k0, lane);
#pragma unroll
    for (int t = 0; t < 4; ++t) {
      v16bf b = load_b_frag(W, DMODEL, n0 + t * 16, k0, lane);
      acc[0][t] = wmma_bf16(a0, b, acc[0][t]);
      acc[1][t] = wmma_bf16(a1, b, acc[1][t]);
    }
  }

  const int N = lane & 15, g = (lane >> 4) & 1;
#pragma unroll
  for (int u = 0; u < 2; ++u)
#pragma unroll
    for (int t = 0; t < 4; ++t) {
      const int e  = n0 + t * 16 + N;
      const int h  = e >> 6;          // head
      const int dc = e & 63;          // dim within head
#pragma unroll
      for (int r = 0; r < 8; ++r) {
        const int n = m0 + u * 16 + r + 8 * g;
        const int b = n >> 12;        // n / 4096
        const int s = n & (S_LEN - 1);
        const size_t bh = (size_t)(b * 8 + h);
        const unsigned short val = f2bf(acc[u][t][r]);
        if (z == 0)      Qws[(bh * S_LEN + s) * DH + dc] = val;
        else if (z == 1) Kws[(bh * S_LEN + s) * DH + dc] = val;
        else             Vtws[(bh * DH + dc) * S_LEN + s] = val;   // transposed
      }
    }
}

// ---------------------------------------------------------------------------
// Kernel 2: flash attention. grid = (S/64, BH), block = 128 (4 waves).
// Each wave owns 16 query rows, streams 64-wide key blocks.
// ---------------------------------------------------------------------------
__global__ __launch_bounds__(128) void flash_attn_kernel(
    const unsigned short* __restrict__ Qws,
    const unsigned short* __restrict__ Kws,
    const unsigned short* __restrict__ Vtws,
    unsigned short* __restrict__ AOws) {
  __shared__ __attribute__((aligned(16))) unsigned short plds[4][16][72];

  const int lane = threadIdx.x & 31;
  const int wave = threadIdx.x >> 5;
  const int bh   = blockIdx.y;
  const int q0   = blockIdx.x * 64 + wave * 16;

  const unsigned short* Qp = Qws + (size_t)bh * S_LEN * DH;
  const unsigned short* Kp = Kws + (size_t)bh * S_LEN * DH;
  const unsigned short* Vp = Vtws + (size_t)bh * DH * S_LEN;

  // Q fragments for the whole tile (K-dim = dh = 64 -> two 16x32 slabs)
  v16bf aq[2];
  aq[0] = load_a_frag(Qp, DH, q0, 0, lane);
  aq[1] = load_a_frag(Qp, DH, q0, 32, lane);

  v8f o[4];
#pragma unroll
  for (int t = 0; t < 4; ++t)
#pragma unroll
    for (int r = 0; r < 8; ++r) o[t][r] = 0.0f;

  float mrun[8], lrun[8];
#pragma unroll
  for (int r = 0; r < 8; ++r) { mrun[r] = -1e30f; lrun[r] = 0.0f; }

  const float scale = 0.125f;  // 1/sqrt(64)
  const int N = lane & 15, g = (lane >> 4) & 1;
  unsigned short* prow = &plds[wave][0][0];

  for (int kb = 0; kb < S_LEN; kb += 64) {
    // prefetch next key/value block toward the caches
    if (kb + 64 < S_LEN) {
      __builtin_prefetch(Kp + (size_t)(kb + 64 + N) * DH, 0, 3);
      __builtin_prefetch(Vp + (size_t)N * S_LEN + kb + 64, 0, 3);
    }

    // ---- scores tile S = (Q K^T) * scale, 16x64 as 4 accumulators ----
    v8f st[4];
#pragma unroll
    for (int t = 0; t < 4; ++t) {
      v8f c;
#pragma unroll
      for (int r = 0; r < 8; ++r) c[r] = 0.0f;
      v16bf b0 = load_b_frag(Kp, DH, kb + t * 16, 0, lane);
      c = wmma_bf16(aq[0], b0, c);
      v16bf b1 = load_b_frag(Kp, DH, kb + t * 16, 32, lane);
      c = wmma_bf16(aq[1], b1, c);
      st[t] = c;
    }

    // ---- online softmax ----
    float bm[8];
#pragma unroll
    for (int r = 0; r < 8; ++r) {
      float v0 = st[0][r] * scale; st[0][r] = v0;
      float v1 = st[1][r] * scale; st[1][r] = v1;
      float v2 = st[2][r] * scale; st[2][r] = v2;
      float v3 = st[3][r] * scale; st[3][r] = v3;
      float m = fmaxf(fmaxf(v0, v1), fmaxf(v2, v3));
      m = fmaxf(m, __shfl_xor(m, 1, 32));
      m = fmaxf(m, __shfl_xor(m, 2, 32));
      m = fmaxf(m, __shfl_xor(m, 4, 32));
      m = fmaxf(m, __shfl_xor(m, 8, 32));
      bm[r] = m;
    }
    float alpha[8], mnew[8], rs[8];
#pragma unroll
    for (int r = 0; r < 8; ++r) {
      mnew[r]  = fmaxf(mrun[r], bm[r]);
      alpha[r] = __expf(mrun[r] - mnew[r]);
      mrun[r]  = mnew[r];
      rs[r]    = 0.0f;
    }
#pragma unroll
    for (int t = 0; t < 4; ++t)
#pragma unroll
      for (int r = 0; r < 8; ++r) {
        float p = __expf(st[t][r] - mnew[r]);
        st[t][r] = p;
        rs[r] += p;
      }
#pragma unroll
    for (int r = 0; r < 8; ++r) {
      float s = rs[r];
      s += __shfl_xor(s, 1, 32);
      s += __shfl_xor(s, 2, 32);
      s += __shfl_xor(s, 4, 32);
      s += __shfl_xor(s, 8, 32);
      lrun[r] = lrun[r] * alpha[r] + s;
    }
#pragma unroll
    for (int t = 0; t < 4; ++t)
#pragma unroll
      for (int r = 0; r < 8; ++r) o[t][r] *= alpha[r];

    // ---- relayout P (C/D layout) -> A fragments via LDS ----
#pragma unroll
    for (int t = 0; t < 4; ++t)
#pragma unroll
      for (int r = 0; r < 8; ++r)
        plds[wave][r + 8 * g][t * 16 + N] = f2bf(st[t][r]);
    __syncthreads();
    v16bf ap[2];
    ap[0] = load_a_frag(prow, 72, 0, 0, lane);
    ap[1] = load_a_frag(prow, 72, 0, 32, lane);
    __syncthreads();

    // ---- O += P @ V  (V transposed: [dh, S]) ----
#pragma unroll
    for (int t = 0; t < 4; ++t) {
      v16bf bv0 = load_b_frag(Vp, S_LEN, t * 16, kb, lane);
      o[t] = wmma_bf16(ap[0], bv0, o[t]);
      v16bf bv1 = load_b_frag(Vp, S_LEN, t * 16, kb + 32, lane);
      o[t] = wmma_bf16(ap[1], bv1, o[t]);
    }
  }

  // ---- normalize and write merged-head output [NROW, DMODEL] bf16 ----
  const int b = bh >> 3, h = bh & 7;
#pragma unroll
  for (int t = 0; t < 4; ++t) {
    const int e = h * DH + t * 16 + N;
#pragma unroll
    for (int r = 0; r < 8; ++r) {
      const int s = q0 + r + 8 * g;
      const size_t n = (size_t)b * S_LEN + s;
      AOws[n * DMODEL + e] = f2bf(o[t][r] / lrun[r]);
    }
  }
}

// ---------------------------------------------------------------------------
// Kernel 3: output projection + bias. grid = (DMODEL/64, NROW/128), block 128.
// Each wave: 32 (M) x 64 (N).  out[n,d] = sum_e AO[n,e] * Wo[d,e] + bo[d]
// ---------------------------------------------------------------------------
__global__ __launch_bounds__(128) void out_proj_kernel(
    const unsigned short* __restrict__ AO,
    const unsigned short* __restrict__ Wo,
    const float* __restrict__ bo,
    float* __restrict__ out) {
  const int lane = threadIdx.x & 31;
  const int wave = threadIdx.x >> 5;
  const int n0   = blockIdx.x * 64;
  const int m0   = blockIdx.y * 128 + wave * 32;

  v8f acc[2][4];
#pragma unroll
  for (int u = 0; u < 2; ++u)
#pragma unroll
    for (int t = 0; t < 4; ++t)
#pragma unroll
      for (int r = 0; r < 8; ++r) acc[u][t][r] = 0.0f;

  for (int k0 = 0; k0 < DMODEL; k0 += 32) {
    v16bf a0 = load_a_frag(AO, DMODEL, m0, k0, lane);
    v16bf a1 = load_a_frag(AO, DMODEL, m0 + 16, k0, lane);
#pragma unroll
    for (int t = 0; t < 4; ++t) {
      v16bf b = load_b_frag(Wo, DMODEL, n0 + t * 16, k0, lane);
      acc[0][t] = wmma_bf16(a0, b, acc[0][t]);
      acc[1][t] = wmma_bf16(a1, b, acc[1][t]);
    }
  }

  const int N = lane & 15, g = (lane >> 4) & 1;
#pragma unroll
  for (int u = 0; u < 2; ++u)
#pragma unroll
    for (int t = 0; t < 4; ++t) {
      const int e = n0 + t * 16 + N;
      const float bias = bo[e];
#pragma unroll
      for (int r = 0; r < 8; ++r) {
        const int n = m0 + u * 16 + r + 8 * g;
        out[(size_t)n * DMODEL + e] = acc[u][t][r] + bias;
      }
    }
}

// ---------------------------------------------------------------------------
extern "C" void kernel_launch(void* const* d_in, const int* in_sizes, int n_in,
                              void* d_out, int out_size, void* d_ws, size_t ws_size,
                              hipStream_t stream) {
  (void)in_sizes; (void)n_in; (void)out_size; (void)ws_size;
  const float* X  = (const float*)d_in[0];
  const float* Wq = (const float*)d_in[1];
  const float* Wk = (const float*)d_in[2];
  const float* Wv = (const float*)d_in[3];
  const float* Wo = (const float*)d_in[4];
  const float* bo = (const float*)d_in[5];
  float* out = (float*)d_out;

  const size_t SZ_X = (size_t)NROW * DMODEL;   // 4,194,304 elems
  const size_t SZ_W = (size_t)DMODEL * DMODEL; //   262,144 elems

  char* ws = (char*)d_ws;
  size_t off = 0;
  unsigned short* Xbf  = (unsigned short*)(ws + off); off += SZ_X * 2;
  unsigned short* Wqbf = (unsigned short*)(ws + off); off += SZ_W * 2;
  unsigned short* Wkbf = (unsigned short*)(ws + off); off += SZ_W * 2;
  unsigned short* Wvbf = (unsigned short*)(ws + off); off += SZ_W * 2;
  unsigned short* Wobf = (unsigned short*)(ws + off); off += SZ_W * 2;
  unsigned short* Qws  = (unsigned short*)(ws + off); off += SZ_X * 2;
  unsigned short* Kws  = (unsigned short*)(ws + off); off += SZ_X * 2;
  unsigned short* Vtws = (unsigned short*)(ws + off); off += SZ_X * 2;
  unsigned short* AOws = (unsigned short*)(ws + off); off += SZ_X * 2;
  // total workspace: ~44 MB

  // 0) downconvert inputs to bf16
  {
    int n4 = (int)(SZ_X / 4);
    cvt_bf16_kernel<<<(n4 + 255) / 256, 256, 0, stream>>>(X, Xbf, n4);
    int w4 = (int)(SZ_W / 4);
    cvt_bf16_kernel<<<(w4 + 255) / 256, 256, 0, stream>>>(Wq, Wqbf, w4);
    cvt_bf16_kernel<<<(w4 + 255) / 256, 256, 0, stream>>>(Wk, Wkbf, w4);
    cvt_bf16_kernel<<<(w4 + 255) / 256, 256, 0, stream>>>(Wv, Wvbf, w4);
    cvt_bf16_kernel<<<(w4 + 255) / 256, 256, 0, stream>>>(Wo, Wobf, w4);
  }

  // 1) QKV projections (z: 0=Q, 1=K, 2=V-transposed)
  qkv_gemm_kernel<<<dim3(DMODEL / 64, NROW / 128, 3), 128, 0, stream>>>(
      Xbf, Wqbf, Wkbf, Wvbf, Qws, Kws, Vtws);

  // 2) flash attention
  flash_attn_kernel<<<dim3(S_LEN / 64, NBH), 128, 0, stream>>>(
      Qws, Kws, Vtws, AOws);

  // 3) output projection + bias
  out_proj_kernel<<<dim3(DMODEL / 64, NROW / 128), 128, 0, stream>>>(
      AOws, Wobf, bo, out);
}
